// TimeLSTM_29489245454840
// MI455X (gfx1250) — matrix operand order repaired and checked by reading, not compiled, not run on hardware
//
#include <hip/hip_runtime.h>
#include <hip/hip_bf16.h>

// ---------------------------------------------------------------------------
// TimeLSTM on gfx1250 (MI455X): bf16 WMMA recurrent steps, f32 accumulate.
// Software-pipelined fragment stream (double-buffered K chunks).
//   B=128, S=512, H=768
// ---------------------------------------------------------------------------

#define TL_B 128
#define TL_S 512
#define TL_H 768

typedef __attribute__((ext_vector_type(16))) __bf16 v16bf;
typedef __attribute__((ext_vector_type(8)))  __bf16 v8bf;
typedef __attribute__((ext_vector_type(8)))  float  v8f;

// ---- fp32 -> bf16 weight conversion (one-time prep) -----------------------
__global__ void tl_cvt_f32_bf16(const float* __restrict__ in,
                                __bf16* __restrict__ out, long n) {
    long i = (long)blockIdx.x * blockDim.x + threadIdx.x;
    if (i < n) out[i] = (__bf16)in[i];
}

// ---- fragment loaders -----------------------------------------------------
// A-matrix 16x32 bf16 layout (ISA 7.12.2): lane L (m=L%16, half=L/16),
// element e<8  -> K = kb + 8*half + e
// element e>=8 -> K = kb + 16 + 8*half + (e-8)
__device__ inline v16bf tl_load_a_f32(const float* __restrict__ row,
                                      int kb, int half) {
    const float4* p0 = (const float4*)(row + kb + 8 * half);
    const float4* p1 = (const float4*)(row + kb + 16 + 8 * half);
    float4 q0 = p0[0], q1 = p0[1];
    float4 q2 = p1[0], q3 = p1[1];
    v16bf a;
    a[0]  = (__bf16)q0.x; a[1]  = (__bf16)q0.y; a[2]  = (__bf16)q0.z; a[3]  = (__bf16)q0.w;
    a[4]  = (__bf16)q1.x; a[5]  = (__bf16)q1.y; a[6]  = (__bf16)q1.z; a[7]  = (__bf16)q1.w;
    a[8]  = (__bf16)q2.x; a[9]  = (__bf16)q2.y; a[10] = (__bf16)q2.z; a[11] = (__bf16)q2.w;
    a[12] = (__bf16)q3.x; a[13] = (__bf16)q3.y; a[14] = (__bf16)q3.z; a[15] = (__bf16)q3.w;
    return a;
}

// B-matrix 32x16 bf16 layout: lane L (n=L%16, half=L/16), element e<8 -> K =
// kb + 8*half + e, e>=8 -> K = kb + 16 + 8*half + (e-8). With W row-major
// [N][K] (B = W^T), a lane's fragment is two b128 loads in row n, no transpose.
__device__ inline v16bf tl_load_b_bf16(const __bf16* __restrict__ rown,
                                       int kb, int half) {
    const v8bf* p = (const v8bf*)(rown + kb + 8 * half);
    v8bf lo = p[0], hi = p[2];
    v16bf b;
#pragma unroll
    for (int e = 0; e < 8; ++e) { b[e] = lo[e]; b[8 + e] = hi[e]; }
    return b;
}

__device__ inline v8f tl_wmma(v16bf a, v16bf b, v8f c) {
    return __builtin_amdgcn_wmma_f32_16x16x32_bf16(
        false, a, false, b, (short)0, c, false, false);
}

__device__ inline float tl_sigmoid(float x) {
    return 1.0f / (1.0f + __expf(-x));
}

// All fragments for one K=32 chunk.
struct TLFrags {
    v16bf aH, aX, aC;
    v16bf bWF, bWI, bWO, bWT;
    v16bf bUF, bUI, bUO, bUT;
    v16bf bWD;
};

__device__ inline void tl_load_chunk(
    TLFrags& f,
    const float* hr, const float* xr, const float* cr,
    const __bf16* wF, const __bf16* wI, const __bf16* wO, const __bf16* wT,
    const __bf16* uF, const __bf16* uI, const __bf16* uO, const __bf16* uT,
    const __bf16* wD, int kb, int half)
{
    f.aH  = tl_load_a_f32(hr, kb, half);
    f.aX  = tl_load_a_f32(xr, kb, half);
    f.aC  = tl_load_a_f32(cr, kb, half);
    f.bWF = tl_load_b_bf16(wF, kb, half);
    f.bWI = tl_load_b_bf16(wI, kb, half);
    f.bWO = tl_load_b_bf16(wO, kb, half);
    f.bWT = tl_load_b_bf16(wT, kb, half);
    f.bUF = tl_load_b_bf16(uF, kb, half);
    f.bUI = tl_load_b_bf16(uI, kb, half);
    f.bUO = tl_load_b_bf16(uO, kb, half);
    f.bUT = tl_load_b_bf16(uT, kb, half);
    f.bWD = tl_load_b_bf16(wD, kb, half);
}

// 9 WMMAs per chunk; same-accumulator pairs separated by 4 independent ops.
__device__ inline void tl_mma_chunk(const TLFrags& f,
                                    v8f& accF, v8f& accI, v8f& accO,
                                    v8f& accT, v8f& accC)
{
    accF = tl_wmma(f.aH, f.bWF, accF);
    accI = tl_wmma(f.aH, f.bWI, accI);
    accO = tl_wmma(f.aH, f.bWO, accO);
    accT = tl_wmma(f.aH, f.bWT, accT);
    accC = tl_wmma(f.aC, f.bWD, accC);
    accF = tl_wmma(f.aX, f.bUF, accF);
    accI = tl_wmma(f.aX, f.bUI, accI);
    accO = tl_wmma(f.aX, f.bUO, accO);
    accT = tl_wmma(f.aX, f.bUT, accT);
}

// ---- one recurrent step ---------------------------------------------------
// grid = (H/16, B/16), block = 32 threads (one wave; EXEC all-ones for WMMA)
__global__ __launch_bounds__(32) void tl_step(
    const float* __restrict__ x,        // inputs + s*H, row stride S*H
    const float* __restrict__ hprev,    // row stride hstride (h0 or d_out slice)
    long hstride,
    const float* __restrict__ ccur,     // [B,H] fp32
    const float* __restrict__ ts,       // timestamps + s, row stride S
    const __bf16* __restrict__ Wall,    // [4H,H] bf16
    const __bf16* __restrict__ Uall,    // [4H,H] bf16
    const __bf16* __restrict__ Wd,      // [H,H]  bf16
    const float* __restrict__ bW,       // [4H]
    const float* __restrict__ bU,       // [4H]
    const float* __restrict__ bd,       // [H]
    float* __restrict__ out_s,          // d_out + s*H, row stride S*H
    float* __restrict__ cnxt)           // [B,H] fp32
{
    const int lane = threadIdx.x;
    const int m    = lane & 15;
    const int half = lane >> 4;
    const int col0 = blockIdx.x * 16;   // column tile within H
    const int row0 = blockIdx.y * 16;   // row tile within B
    const int colg = col0 + m;          // B/C column for this lane

    const long SH = (long)TL_S * TL_H;
    const int  bm = row0 + m;           // A-matrix row for this lane

    const float* xr = x     + (long)bm * SH;
    const float* hr = hprev + (long)bm * hstride;
    const float* cr = ccur  + (long)bm * TL_H;

    const __bf16* wF = Wall + (long)(0 * TL_H + colg) * TL_H;
    const __bf16* wI = Wall + (long)(1 * TL_H + colg) * TL_H;
    const __bf16* wO = Wall + (long)(2 * TL_H + colg) * TL_H;
    const __bf16* wT = Wall + (long)(3 * TL_H + colg) * TL_H;
    const __bf16* uF = Uall + (long)(0 * TL_H + colg) * TL_H;
    const __bf16* uI = Uall + (long)(1 * TL_H + colg) * TL_H;
    const __bf16* uO = Uall + (long)(2 * TL_H + colg) * TL_H;
    const __bf16* uT = Uall + (long)(3 * TL_H + colg) * TL_H;
    const __bf16* wD = Wd   + (long)colg * TL_H;

    v8f accF = {}, accI = {}, accO = {}, accT = {}, accC = {};

    // Prologue: first chunk into buffer 0.
    TLFrags f0, f1;
    tl_load_chunk(f0, hr, xr, cr, wF, wI, wO, wT, uF, uI, uO, uT, wD, 0, half);

    // Hoist epilogue operands so their latency hides under the GEMM.
    const float bWf = bW[colg],            bUf = bU[colg];
    const float bWi = bW[TL_H + colg],     bUi = bU[TL_H + colg];
    const float bWo = bW[2 * TL_H + colg], bUo = bU[2 * TL_H + colg];
    const float bWt = bW[3 * TL_H + colg], bUt = bU[3 * TL_H + colg];
    const float bDd = bd[colg];
    float tv[8], cp[8];
#pragma unroll
    for (int r = 0; r < 8; ++r) {
        const int b = row0 + r + 8 * half;
        tv[r] = ts[(long)b * TL_S];
        cp[r] = ccur[(long)b * TL_H + colg];
    }

    // Pipelined main loop: prefetch chunk k+32 while multiplying chunk k.
#pragma unroll 2
    for (int kb = 0; kb < TL_H; kb += 64) {
        // Distance-2 prefetch on the weight streams (-> global_prefetch_b8).
        if (kb + 128 < TL_H) {
            const int kp = kb + 128 + 16 * half;
            __builtin_prefetch(wF + kp, 0, 3);
            __builtin_prefetch(wI + kp, 0, 3);
            __builtin_prefetch(wO + kp, 0, 3);
            __builtin_prefetch(wT + kp, 0, 3);
            __builtin_prefetch(uF + kp, 0, 3);
            __builtin_prefetch(uI + kp, 0, 3);
            __builtin_prefetch(uO + kp, 0, 3);
            __builtin_prefetch(uT + kp, 0, 3);
            __builtin_prefetch(wD + kp, 0, 3);
        }
        tl_load_chunk(f1, hr, xr, cr, wF, wI, wO, wT, uF, uI, uO, uT, wD,
                      kb + 32, half);
        tl_mma_chunk(f0, accF, accI, accO, accT, accC);
        if (kb + 64 < TL_H)
            tl_load_chunk(f0, hr, xr, cr, wF, wI, wO, wT, uF, uI, uO, uT, wD,
                          kb + 64, half);
        tl_mma_chunk(f1, accF, accI, accO, accT, accC);
    }

    // Epilogue. C/D layout: VGPR r holds row (r + 8*half), col = lane%16.
#pragma unroll
    for (int r = 0; r < 8; ++r) {
        const int b = row0 + r + 8 * half;

        const float cs1  = tanhf(accC[r] + bDd);
        const float cadj = (cp[r] - cs1) + cs1 * tv[r];

        const float fg = tl_sigmoid(accF[r] + bWf + bUf);
        const float ig = tl_sigmoid(accI[r] + bWi + bUi);
        const float og = tl_sigmoid(accO[r] + bWo + bUo);
        const float ct = tl_sigmoid(accT[r] + bWt + bUt);

        const float cn = fg * cadj + ig * ct;
        const float hn = og * tanhf(cn);

        out_s[(long)b * SH + colg]  = hn;
        cnxt[(long)b * TL_H + colg] = cn;
    }
}

// ---------------------------------------------------------------------------
extern "C" void kernel_launch(void* const* d_in, const int* in_sizes, int n_in,
                              void* d_out, int out_size, void* d_ws, size_t ws_size,
                              hipStream_t stream) {
    (void)in_sizes; (void)n_in; (void)out_size; (void)ws_size;
    constexpr int B = TL_B, S = TL_S, H = TL_H;

    const float* inputs = (const float*)d_in[0];   // [B,S,H]
    const float* tstamp = (const float*)d_in[1];   // [B,S]
    const float* h0     = (const float*)d_in[2];   // [B,H]
    const float* c0     = (const float*)d_in[3];   // [B,H]
    const float* Wall_w = (const float*)d_in[4];   // [4H,H]
    const float* Wall_b = (const float*)d_in[5];   // [4H]
    const float* Uall_w = (const float*)d_in[6];   // [4H,H]
    const float* Uall_b = (const float*)d_in[7];   // [4H]
    const float* Wd_w   = (const float*)d_in[8];   // [H,H]
    const float* Wd_b   = (const float*)d_in[9];   // [H]
    float* out = (float*)d_out;                    // [B,S,H]

    // Workspace carve (256B aligned): bf16 weights + fp32 c ping-pong.
    char*  ws  = (char*)d_ws;
    size_t off = 0;
    auto carve = [&](size_t bytes) -> void* {
        void* p = ws + off;
        off = (off + bytes + 255) & ~(size_t)255;
        return p;
    };
    __bf16* Wall_bf = (__bf16*)carve((size_t)4 * H * H * sizeof(__bf16));
    __bf16* Uall_bf = (__bf16*)carve((size_t)4 * H * H * sizeof(__bf16));
    __bf16* Wd_bf   = (__bf16*)carve((size_t)H * H * sizeof(__bf16));
    float*  cbuf0   = (float*)carve((size_t)B * H * sizeof(float));
    float*  cbuf1   = (float*)carve((size_t)B * H * sizeof(float));
    float*  cbuf[2] = { cbuf0, cbuf1 };

    // Prep: convert weights to bf16, seed c ping-pong with c0.
    {
        long nw = (long)4 * H * H;
        tl_cvt_f32_bf16<<<(unsigned)((nw + 255) / 256), 256, 0, stream>>>(Wall_w, Wall_bf, nw);
        tl_cvt_f32_bf16<<<(unsigned)((nw + 255) / 256), 256, 0, stream>>>(Uall_w, Uall_bf, nw);
        long nd = (long)H * H;
        tl_cvt_f32_bf16<<<(unsigned)((nd + 255) / 256), 256, 0, stream>>>(Wd_w, Wd_bf, nd);
        hipMemcpyAsync(cbuf[0], c0, (size_t)B * H * sizeof(float),
                       hipMemcpyDeviceToDevice, stream);
    }

    // Sequential scan: one launch per timestep (graph replay amortizes this).
    const dim3 grid(H / 16, B / 16);
    for (int s = 0; s < S; ++s) {
        const float* hprev   = (s == 0) ? h0 : (out + (size_t)(s - 1) * H);
        const long   hstride = (s == 0) ? (long)H : (long)S * H;
        const float* ccur    = cbuf[s & 1];
        float*       cnxt    = cbuf[(s & 1) ^ 1];
        tl_step<<<grid, 32, 0, stream>>>(
            inputs + (size_t)s * H, hprev, hstride, ccur, tstamp + s,
            Wall_bf, Uall_bf, Wd_bf, Wall_b, Uall_b, Wd_b,
            out + (size_t)s * H, cnxt);
    }
}
